// MultiHeadAttention_49667001810987
// MI455X (gfx1250) — compile-verified, matrix operand order checked by reading, not run on hardware
//
#include <hip/hip_runtime.h>
#include <hip/hip_bf16.h>

// ---------------------------------------------------------------------------
// MI455X (gfx1250, wave32) fused multi-head attention:
//   proj_qkv (WMMA f16, V stored transposed) -> flash-attn (WMMA f16 +
//   LDS transpose + shfl) -> proj_out (WMMA f16)
// ---------------------------------------------------------------------------

typedef __attribute__((ext_vector_type(16))) _Float16 v16h;
typedef __attribute__((ext_vector_type(8)))  _Float16 v8h;
typedef __attribute__((ext_vector_type(8)))  float    v8f;

#define BS 4
#define NN 2048
#define DD 512
#define HH 8
#define DK 64

static __device__ __forceinline__ v8f v8f_zero() {
    v8f z;
#pragma unroll
    for (int i = 0; i < 8; ++i) z[i] = 0.0f;
    return z;
}

// Load a 16-element f16 fragment (A-row / B-col pattern) from an f32 row:
// elems 0..7 <- p[0..7], elems 8..15 <- p[16..23].  p is 32B aligned.
static __device__ __forceinline__ v16h load16_f32(const float* __restrict__ p) {
    const float4* q = (const float4*)p;
    float4 f0 = q[0], f1 = q[1], f2 = q[4], f3 = q[5];
    v16h r;
    r[0]  = (_Float16)f0.x; r[1]  = (_Float16)f0.y; r[2]  = (_Float16)f0.z; r[3]  = (_Float16)f0.w;
    r[4]  = (_Float16)f1.x; r[5]  = (_Float16)f1.y; r[6]  = (_Float16)f1.z; r[7]  = (_Float16)f1.w;
    r[8]  = (_Float16)f2.x; r[9]  = (_Float16)f2.y; r[10] = (_Float16)f2.z; r[11] = (_Float16)f2.w;
    r[12] = (_Float16)f3.x; r[13] = (_Float16)f3.y; r[14] = (_Float16)f3.z; r[15] = (_Float16)f3.w;
    return r;
}

// Same pattern from an f16 row (two 16B vector loads).
static __device__ __forceinline__ v16h load16_f16(const _Float16* __restrict__ p) {
    v8h lo = *(const v8h*)p;
    v8h hi = *(const v8h*)(p + 16);
    v16h r;
#pragma unroll
    for (int i = 0; i < 8; ++i) { r[i] = lo[i]; r[i + 8] = hi[i]; }
    return r;
}

static __device__ __forceinline__ v8f wmma_f16(v16h a, v16h b, v8f c) {
    return __builtin_amdgcn_wmma_f32_16x16x32_f16(false, a, false, b, (short)0, c, false, false);
}

// ---------------------------------------------------------------------------
// Kernel 1: projections, f32 in -> f16 out.
//   Qh, Kh : mask * (X @ W^T), head-major (b,h,n,dk)   [rows contiguous]
//   Vt     : mask * (V @ WV^T), TRANSPOSED (b,h,dk,n)  [dk-cols contiguous]
// One wave computes a 16-row x 64-col tile. grid = (8 strips==heads, 512 row
// tiles, 3 matrices), block = 32.
// ---------------------------------------------------------------------------
__global__ __launch_bounds__(32) void proj_qkv_kernel(
    const float* __restrict__ Q, const float* __restrict__ K, const float* __restrict__ V,
    const float* __restrict__ WQ, const float* __restrict__ WK, const float* __restrict__ WV,
    const float* __restrict__ qm, const float* __restrict__ km,
    _Float16* __restrict__ Qh, _Float16* __restrict__ Kh, _Float16* __restrict__ Vt)
{
    const int lane  = threadIdx.x;
    const int lmod  = lane & 15;
    const int hi8   = (lane >> 4) << 3;   // 0 or 8
    const int strip = blockIdx.x;         // 64-col strip == head index
    const int rowt  = blockIdx.y;         // 16-row tile over BS*N
    const int which = blockIdx.z;

    const float* X   = (which == 0) ? Q  : (which == 1) ? K  : V;
    const float* W   = (which == 0) ? WQ : (which == 1) ? WK : WV;
    const float* msk = (which == 0) ? qm : km;

    const float* xrow = X + (size_t)(rowt * 16 + lmod) * DD;

    v8f acc[4];
#pragma unroll
    for (int j = 0; j < 4; ++j) acc[j] = v8f_zero();

    for (int kc = 0; kc < DD; kc += 32) {
        v16h a = load16_f32(xrow + kc + hi8);
#pragma unroll
        for (int j = 0; j < 4; ++j) {
            const int wrow = strip * 64 + j * 16 + lmod;
            v16h b = load16_f32(W + (size_t)wrow * DD + kc + hi8);
            acc[j] = wmma_f16(a, b, acc[j]);
        }
    }

    // D layout: row = elem + hi8 (+tile base), col = lmod (+16*j + 64*strip)
    const int b_ = (rowt * 16) >> 11;           // batch (tile never crosses)
    const int n0 = (rowt * 16) & (NN - 1);      // n of row 0 of tile

    if (which == 2) {
        // V: transposed store (b,h,dk,n) -- rows r are contiguous in n.
#pragma unroll
        for (int j = 0; j < 4; ++j) {
            const int dk = j * 16 + lmod;
            v8h pk;
#pragma unroll
            for (int r = 0; r < 8; ++r)
                pk[r] = (_Float16)(acc[j][r] * msk[rowt * 16 + r + hi8]);
            *(v8h*)&Vt[(((size_t)b_ * HH + strip) * DK + dk) * NN + n0 + hi8] = pk;
        }
    } else {
        _Float16* out = (which == 0) ? Qh : Kh;
#pragma unroll
        for (int j = 0; j < 4; ++j) {
            const int dk = j * 16 + lmod;
#pragma unroll
            for (int r = 0; r < 8; ++r) {
                const int grow = rowt * 16 + r + hi8;   // over BS*N
                const float v  = acc[j][r] * msk[grow];
                out[(((size_t)b_ * HH + strip) * NN + (n0 + r + hi8)) * DK + dk] =
                    (_Float16)v;
            }
        }
    }
}

// ---------------------------------------------------------------------------
// Kernel 2: flash attention. One wave per (b, h, 16-query tile).
// grid = (N/16, H, BS), block = 32.
// ---------------------------------------------------------------------------
__global__ __launch_bounds__(32) void attn_kernel(
    const _Float16* __restrict__ Qh, const _Float16* __restrict__ Kh,
    const _Float16* __restrict__ Vt, const float* __restrict__ att,
    const float* __restrict__ km, _Float16* __restrict__ Yw)
{
    __shared__ __align__(16) _Float16 Pl[16 * 32];   // P tile: 16 rows x 32 keys

    const int lane = threadIdx.x;
    const int lmod = lane & 15;
    const int hi8  = (lane >> 4) << 3;
    const int qt   = blockIdx.x;
    const int h    = blockIdx.y;
    const int b    = blockIdx.z;

    const size_t headbase = ((size_t)b * HH + h) * NN * DK;   // same for Qh/Kh/Vt
    const int q0 = qt * 16;

    // Q fragment (16 rows x K=64, as two K=32 A fragments), resident all loop.
    const _Float16* qrow = Qh + headbase + (size_t)(q0 + lmod) * DK;
    const v16h aq0 = load16_f16(qrow + hi8);
    const v16h aq1 = load16_f16(qrow + 32 + hi8);

    v8f acc[4];
#pragma unroll
    for (int j = 0; j < 4; ++j) acc[j] = v8f_zero();
    float mrun[8], lrun[8];
#pragma unroll
    for (int r = 0; r < 8; ++r) { mrun[r] = -1e30f; lrun[r] = 0.0f; }

    const float* attb = att + ((size_t)b * NN + q0) * NN;

    for (int kk = 0; kk < NN; kk += 32) {
        // prefetch next mask block into the cache hierarchy
        if (kk + 32 < NN)
            __builtin_prefetch(attb + (size_t)hi8 * NN + kk + 32, 0, 0);

        // ---- scores for two 16-key tiles (4 WMMAs) ----
        v8f s[2];
#pragma unroll
        for (int t = 0; t < 2; ++t) {
            const _Float16* krow = Kh + headbase + (size_t)(kk + t * 16 + lmod) * DK;
            v16h bk0 = load16_f16(krow + hi8);
            v16h bk1 = load16_f16(krow + 32 + hi8);
            v8f sc = v8f_zero();
            sc = wmma_f16(aq0, bk0, sc);
            sc = wmma_f16(aq1, bk1, sc);
            const int key = kk + t * 16 + lmod;
#pragma unroll
            for (int r = 0; r < 8; ++r) {
                const float m = attb[(size_t)(r + hi8) * NN + key];
                sc[r] = (m != 0.0f) ? sc[r] * 0.125f : -1e30f;   // 1/sqrt(64)
            }
            s[t] = sc;
        }

        // ---- online softmax: row max over 32 keys (reduce across 16 lanes) ----
        float mt[8];
#pragma unroll
        for (int r = 0; r < 8; ++r) mt[r] = fmaxf(s[0][r], s[1][r]);
#pragma unroll
        for (int mk = 1; mk < 16; mk <<= 1)
#pragma unroll
            for (int r = 0; r < 8; ++r) mt[r] = fmaxf(mt[r], __shfl_xor(mt[r], mk, 32));

        float scale_[8];
#pragma unroll
        for (int r = 0; r < 8; ++r) {
            const float mn = fmaxf(mrun[r], mt[r]);
            scale_[r] = __expf(mrun[r] - mn);
            mrun[r] = mn;
        }

        // ---- P = exp(s - m), partial row sums, stage P to LDS (C -> A) ----
        float ls[8];
#pragma unroll
        for (int r = 0; r < 8; ++r) ls[r] = 0.0f;
#pragma unroll
        for (int t = 0; t < 2; ++t)
#pragma unroll
            for (int r = 0; r < 8; ++r) {
                const float p = __expf(s[t][r] - mrun[r]);
                ls[r] += p;
                Pl[(r + hi8) * 32 + t * 16 + lmod] = (_Float16)p;
            }
#pragma unroll
        for (int mk = 1; mk < 16; mk <<= 1)
#pragma unroll
            for (int r = 0; r < 8; ++r) ls[r] += __shfl_xor(ls[r], mk, 32);
#pragma unroll
        for (int r = 0; r < 8; ++r) lrun[r] = lrun[r] * scale_[r] + ls[r];

        __syncthreads();   // single-wave block: emits LDS waits

        // Re-read P as an A fragment (16 x 32).
        const v16h ap = load16_f16(&Pl[lmod * 32 + hi8]);

        // ---- rescale accumulator, then P x V (4 WMMAs) ----
#pragma unroll
        for (int j = 0; j < 4; ++j)
#pragma unroll
            for (int r = 0; r < 8; ++r) acc[j][r] *= scale_[r];

#pragma unroll
        for (int j = 0; j < 4; ++j) {
            // Vt is (b,h,dk,n): B elem (kidx, dkcol) = Vt[dkcol*NN + kk + kidx]
            const _Float16* vcol = Vt + headbase + (size_t)(j * 16 + lmod) * NN + kk;
            v16h bv = load16_f16(vcol + hi8);
            acc[j] = wmma_f16(ap, bv, acc[j]);
        }
        __syncthreads();
    }

    // ---- epilogue: y = acc / l * km[q] ; store (b, n, d) f16 ----
#pragma unroll
    for (int r = 0; r < 8; ++r) {
        const int qg = q0 + r + hi8;
        const float sc = km[b * NN + qg] / lrun[r];
#pragma unroll
        for (int j = 0; j < 4; ++j) {
            const int col = h * DK + j * 16 + lmod;
            Yw[((size_t)b * NN + qg) * DD + col] = (_Float16)(acc[j][r] * sc);
        }
    }
}

// ---------------------------------------------------------------------------
// Kernel 3: out = (Y @ WO^T) * k_mas, f16 x f32 -> f32.
// grid = (8, 512), block = 32.
// ---------------------------------------------------------------------------
__global__ __launch_bounds__(32) void proj_out_kernel(
    const _Float16* __restrict__ Yw, const float* __restrict__ WO,
    const float* __restrict__ km, float* __restrict__ out)
{
    const int lane  = threadIdx.x;
    const int lmod  = lane & 15;
    const int hi8   = (lane >> 4) << 3;
    const int strip = blockIdx.x;
    const int rowt  = blockIdx.y;

    const _Float16* yrow = Yw + (size_t)(rowt * 16 + lmod) * DD;

    v8f acc[4];
#pragma unroll
    for (int j = 0; j < 4; ++j) acc[j] = v8f_zero();

    for (int kc = 0; kc < DD; kc += 32) {
        v16h a = load16_f16(yrow + kc + hi8);
#pragma unroll
        for (int j = 0; j < 4; ++j) {
            const int wrow = strip * 64 + j * 16 + lmod;
            v16h b = load16_f32(WO + (size_t)wrow * DD + kc + hi8);
            acc[j] = wmma_f16(a, b, acc[j]);
        }
    }

#pragma unroll
    for (int j = 0; j < 4; ++j) {
        const int col = strip * 64 + j * 16 + lmod;
#pragma unroll
        for (int r = 0; r < 8; ++r) {
            const int grow = rowt * 16 + r + hi8;   // over BS*N
            out[(size_t)grow * DD + col] = acc[j][r] * km[grow];
        }
    }
}

// ---------------------------------------------------------------------------
extern "C" void kernel_launch(void* const* d_in, const int* in_sizes, int n_in,
                              void* d_out, int out_size, void* d_ws, size_t ws_size,
                              hipStream_t stream) {
    const float* Q   = (const float*)d_in[0];
    const float* K   = (const float*)d_in[1];
    const float* V   = (const float*)d_in[2];
    const float* qm  = (const float*)d_in[3];
    const float* km  = (const float*)d_in[4];
    const float* att = (const float*)d_in[5];
    const float* WQ  = (const float*)d_in[6];
    const float* WK  = (const float*)d_in[7];
    const float* WV  = (const float*)d_in[8];
    const float* WO  = (const float*)d_in[9];
    float* out = (float*)d_out;

    const size_t SZ = (size_t)BS * NN * DD;   // elements per f16 tensor
    _Float16* Qh = (_Float16*)d_ws;
    _Float16* Kh = Qh + SZ;
    _Float16* Vt = Kh + SZ;   // transposed (b,h,dk,n)
    _Float16* Yw = Vt + SZ;

    // 1) projections: Qh/Kh (head-major), Vt (transposed), all f16
    proj_qkv_kernel<<<dim3(HH, (BS * NN) / 16, 3), dim3(32), 0, stream>>>(
        Q, K, V, WQ, WK, WV, qm, km, Qh, Kh, Vt);

    // 2) flash attention -> Yw (f16, (b,n,d))
    attn_kernel<<<dim3(NN / 16, HH, BS), dim3(32), 0, stream>>>(
        Qh, Kh, Vt, att, km, Yw);

    // 3) output projection -> d_out (f32)
    proj_out_kernel<<<dim3(HH, (BS * NN) / 16), dim3(32), 0, stream>>>(
        Yw, WO, km, out);
}
